// MulitmodalCompactBilinearPool_88665304859454
// MI455X (gfx1250) — compile-verified, matrix operand order checked by reading, not run on hardware
//
#include <hip/hip_runtime.h>

// Multimodal Compact Bilinear Pooling, MI455X (gfx1250, wave32).
//
// phi[b, (i1[d1]+i2[d2]) % P] += s1*x1[b,d1] * s2*x2[b,d2]
//   K1: extract (idx, sign) per row of C1/C2   (262 MB NT stream = 11.3 us floor)
//   K2: zero the output
//   K3: 16x16 outer-product tiles on the matrix pipe (v_wmma_f32_16x16x32_f16,
//       K=0 slice), scatter via ds_add_f32 into a 2P-long (128 KB) LDS
//       accumulator (no modulo), fold + global_atomic_add_f32 merge.
//       Each wave's 4 tn-tiles are invariant over tm: B operands and i2
//       offsets are hoisted out of the tm loop; j-loop fully unrolled.

#define B_   128
#define D_   2048
#define P_   16000

typedef _Float16 v16h __attribute__((ext_vector_type(16)));
typedef float    v8f  __attribute__((ext_vector_type(8)));
typedef float    f4   __attribute__((ext_vector_type(4)));
typedef int      i4   __attribute__((ext_vector_type(4)));

// ---------------- K1: count-sketch structure extraction -------------------
__global__ __launch_bounds__(256) void mcb_extract(const float* __restrict__ C1,
                                                   const float* __restrict__ C2,
                                                   int*   __restrict__ gidx,
                                                   float* __restrict__ gsgn) {
  const int bid = blockIdx.x;
  const int m   = bid >> 11;           // D_ == 2^11
  const int row = bid & (D_ - 1);
  const float* __restrict__ C = m ? C2 : C1;
  const f4* __restrict__ rp = (const f4*)(C + (size_t)row * P_);
  int*   oi = gidx + m * D_ + row;
  float* os = gsgn + m * D_ + row;
  for (int q = threadIdx.x; q < P_ / 4; q += 256) {
    f4 v = __builtin_nontemporal_load(rp + q);
    if (v.x != 0.f) { *oi = 4 * q + 0; *os = v.x; }
    if (v.y != 0.f) { *oi = 4 * q + 1; *os = v.y; }
    if (v.z != 0.f) { *oi = 4 * q + 2; *os = v.z; }
    if (v.w != 0.f) { *oi = 4 * q + 3; *os = v.w; }
  }
}

// ---------------- K2: zero output (merge target for K3 partials) ----------
__global__ __launch_bounds__(1024) void mcb_zero(float* __restrict__ out) {
  out[(size_t)blockIdx.x * 1024 + threadIdx.x] = 0.f;   // grid 2000*1024 == B_*P_
}

// ---------------- K3: WMMA outer-product + LDS scatter --------------------
#define SPLIT    2
#define THREADS  1024
#define WAVES    (THREADS / 32)
#define MT       (D_ / 16)          // 128 tiles per dim
#define TMB      (MT / SPLIT)       // 64 d1-tiles per block
#define JTN      (MT / WAVES)       // 4 tn-tiles per wave (invariant over tm)
#define PHI2     (2 * P_)           // unwrapped accumulator length (floats)

__global__ __launch_bounds__(THREADS) void mcb_conv(
    const float* __restrict__ x1, const float* __restrict__ x2,
    const int* __restrict__ gidx, const float* __restrict__ gsgn,
    float* __restrict__ out) {
  extern __shared__ char smem[];
  float*    phi2 = (float*)smem;                 // 2*P_ floats (128 KB, no-mod scatter)
  int*      i1s  = (int*)(phi2 + PHI2);          // D_ ints (BYTE offsets, idx*4)
  int*      i2s  = i1s + D_;                     // D_ ints (BYTE offsets)
  _Float16* v1h  = (_Float16*)(i2s + D_);        // D_ halfs
  _Float16* v2h  = v1h + D_;                     // D_ halfs
  // total: 128000 + 8192 + 8192 + 4096 + 4096 = 152576 B  -> 2 blocks / WGP

  const int b     = blockIdx.x >> 1;
  const int split = blockIdx.x & (SPLIT - 1);
  const int tid   = threadIdx.x;

  // Stage sketch values (f16) and byte-offset targets; zero the accumulator.
  for (int d = tid; d < D_; d += THREADS) {
    i1s[d] = gidx[d] << 2;
    i2s[d] = gidx[D_ + d] << 2;
    v1h[d] = (_Float16)(gsgn[d]      * x1[(size_t)b * D_ + d]);
    v2h[d] = (_Float16)(gsgn[D_ + d] * x2[(size_t)b * D_ + d]);
  }
  for (int k = tid; k < PHI2; k += THREADS) phi2[k] = 0.f;
  __syncthreads();

  const int wave   = tid >> 5;
  const int lane   = tid & 31;
  const int lane15 = lane & 15;
  const int hi     = lane >> 4;

  // Per-wave invariants: the 4 tn tiles this wave ever touches. Pre-mask the
  // B-operand halves and fetch the i2 byte-offsets once.
  _Float16 bm[JTN];
  int      i2v[JTN];
#pragma unroll
  for (int j = 0; j < JTN; ++j) {
    const int tn = wave + j * WAVES;
    _Float16 bh = v2h[tn * 16 + lane15];
    bm[j]  = hi ? (_Float16)0.f : bh;
    i2v[j] = i2s[tn * 16 + lane15];
  }

  // A/B live in registers; only element 0 (lane 0-15, K=0 slot) ever changes.
  v16h A  = {};
  v16h Bv = {};

  const int tm0 = split * TMB;
  for (int tm = tm0; tm < tm0 + TMB; ++tm) {     // uniform scalar loop
    _Float16 ah = v1h[tm * 16 + lane15];
    A[0] = hi ? (_Float16)0.f : ah;
    const i4* i1p = (const i4*)(i1s + tm * 16 + (hi << 3));  // 32B-aligned
    const i4 ia = i1p[0];
    const i4 ib = i1p[1];
    char* base = (char*)phi2;

#pragma unroll
    for (int j = 0; j < JTN; ++j) {
      Bv[0] = bm[j];                             // single 16-bit reg write
      v8f c = {};
      // Rank-1 tile on the matrix pipe; EXEC is all-ones (uniform control).
      c = __builtin_amdgcn_wmma_f32_16x16x32_f16(
          false, A, false, Bv, (short)0, c, false, false);

      // Scatter: byte addr = i1b + i2b in [0, 2*64000); no modulo needed.
      const int o = i2v[j];
      atomicAdd((float*)(base + (ia.x + o)), c[0]);
      atomicAdd((float*)(base + (ia.y + o)), c[1]);
      atomicAdd((float*)(base + (ia.z + o)), c[2]);
      atomicAdd((float*)(base + (ia.w + o)), c[3]);
      atomicAdd((float*)(base + (ib.x + o)), c[4]);
      atomicAdd((float*)(base + (ib.y + o)), c[5]);
      atomicAdd((float*)(base + (ib.z + o)), c[6]);
      atomicAdd((float*)(base + (ib.w + o)), c[7]);
    }
  }
  __syncthreads();

  // Fold the unwrapped halves and merge into global output.
  float* orow = out + (size_t)b * P_;
  for (int k = tid; k < P_; k += THREADS)
    unsafeAtomicAdd(&orow[k], phi2[k] + phi2[k + P_]);   // global_atomic_add_f32
}

extern "C" void kernel_launch(void* const* d_in, const int* in_sizes, int n_in,
                              void* d_out, int out_size, void* d_ws, size_t ws_size,
                              hipStream_t stream) {
  const float* x1 = (const float*)d_in[0];   // [128,2048]
  const float* x2 = (const float*)d_in[1];   // [128,2048]
  const float* C1 = (const float*)d_in[2];   // [2048,16000]
  const float* C2 = (const float*)d_in[3];   // [2048,16000]
  float* out = (float*)d_out;                // [128,16000]

  int*   gidx = (int*)d_ws;                              // 2*D_ ints
  float* gsgn = (float*)((char*)d_ws + 2 * D_ * sizeof(int));

  mcb_extract<<<dim3(2 * D_), dim3(256), 0, stream>>>(C1, C2, gidx, gsgn);
  mcb_zero<<<dim3((B_ * P_) / 1024), dim3(1024), 0, stream>>>(out);

  const size_t shmem = (size_t)PHI2 * sizeof(float)
                     + (size_t)(2 * D_) * sizeof(int)
                     + (size_t)(2 * D_) * sizeof(_Float16);   // 152576 B
  (void)hipFuncSetAttribute(reinterpret_cast<const void*>(mcb_conv),
                            hipFuncAttributeMaxDynamicSharedMemorySize,
                            (int)shmem);
  mcb_conv<<<dim3(B_ * SPLIT), dim3(THREADS), shmem, stream>>>(x1, x2, gidx, gsgn, out);
}
